// Attention_41085657153620
// MI455X (gfx1250) — matrix operand order repaired and checked by reading, not compiled, not run on hardware
//
#include <hip/hip_runtime.h>

typedef __attribute__((ext_vector_type(16))) __bf16 v16bf;
typedef __attribute__((ext_vector_type(8)))  float  v8f;

union FragAB { v16bf v; uint4 q[2]; };
union U8     { uint4 v; unsigned short s[8]; };

__device__ inline unsigned short f2bf(float f) {
  unsigned int u = __float_as_uint(f);
  u += 0x7FFFu + ((u >> 16) & 1u);           // round-to-nearest-even
  return (unsigned short)(u >> 16);
}

// async copy: 16B global -> LDS, tracked by ASYNCcnt (CDNA5)
__device__ inline void async_cp16(unsigned ldsOff, const void* g) {
  unsigned long long ga = (unsigned long long)g;
  asm volatile("global_load_async_to_lds_b128 %0, %1, off"
               :: "v"(ldsOff), "v"(ga) : "memory");
}
__device__ inline void async_wait0() {
  asm volatile("s_wait_asynccnt 0x0" ::: "memory");
}

#define DD    512
#define QL    4096
#define KLEN  4096
#define BATCH 8
#define NT    (KLEN / 32)

#define PQ 520   // Qs/Kn row pitch (ushorts): 1040B, 16B-aligned, conflict-free
#define PT 40    // Kt/Pt row pitch (ushorts): 80B, 16B-aligned, conflict-free
#define PS 33    // Ssc row pitch (floats): stride coprime with 64 banks

// ---------------- Kernel 1: prep  ----------------
// Qh = bf16(q * w3), Kh = bf16(k), ek[j] = k_j . w2   (eq_i and b cancel in softmax)
__global__ __launch_bounds__(256) void prep_kernel(
    const float* __restrict__ q, const float* __restrict__ k,
    const float* __restrict__ W,
    unsigned short* __restrict__ Qh, unsigned short* __restrict__ Kh,
    float* __restrict__ ekw)
{
  const int lane = threadIdx.x & 31;
  const int wave = threadIdx.x >> 5;
  if (blockIdx.x < 4096) {
    const size_t row = (size_t)blockIdx.x * 8 + wave;      // q row
    const float4* src = (const float4*)(q + row * DD);
    const float4* w3  = (const float4*)(W + 2 * DD);
    ushort4* dst = (ushort4*)(Qh + row * DD);
    #pragma unroll
    for (int t = 0; t < 4; ++t) {
      const int c = lane + t * 32;
      float4 v = src[c];
      float4 w = w3[c];
      ushort4 o;
      o.x = f2bf(v.x * w.x); o.y = f2bf(v.y * w.y);
      o.z = f2bf(v.z * w.z); o.w = f2bf(v.w * w.w);
      dst[c] = o;
    }
  } else {
    const size_t row = (size_t)(blockIdx.x - 4096) * 8 + wave;  // k row
    const float4* src = (const float4*)(k + row * DD);
    const float4* w2  = (const float4*)(W + DD);
    ushort4* dst = (ushort4*)(Kh + row * DD);
    float acc = 0.0f;
    #pragma unroll
    for (int t = 0; t < 4; ++t) {
      const int c = lane + t * 32;
      float4 v = src[c];
      float4 w = w2[c];
      acc += v.x * w.x + v.y * w.y + v.z * w.z + v.w * w.w;
      ushort4 o;
      o.x = f2bf(v.x); o.y = f2bf(v.y); o.z = f2bf(v.z); o.w = f2bf(v.w);
      dst[c] = o;
    }
    #pragma unroll
    for (int off = 16; off > 0; off >>= 1) acc += __shfl_xor(acc, off);
    if (lane == 0) ekw[row] = acc;
  }
}

// ---------------- Kernel 2: flash attention ----------------
// Block = 256 threads (8 waves). Block owns a 32-row Q tile; streams 32-key tiles
// with double-buffered async global->LDS staging of K.
__global__ __launch_bounds__(256) void flash_attn_kernel(
    const unsigned short* __restrict__ Qh,
    const unsigned short* __restrict__ Kh,
    const float* __restrict__ ekw,
    float* __restrict__ out)
{
  __shared__ unsigned short Qs[32 * PQ];       // (m,d) bf16  ~32.5 KB
  __shared__ unsigned short Kn[2][32 * PQ];    // (n,d) bf16  ~65 KB (double-buffered)
  __shared__ unsigned short Kt[DD * PT];       // (d,n) bf16  40 KB
  __shared__ float          Ssc0[32 * PS];     // partial scores (d 0..255)
  __shared__ float          Ssc1[32 * PS];     // partial scores (d 256..511)
  __shared__ unsigned short Pt[32 * PT];       // probs bf16  2.5 KB
  __shared__ float ekt[2][32];
  __shared__ float rowM[32], rowL[32], rowA[32];

  const int tid  = threadIdx.x;
  const int lane = tid & 31;
  const int wave = tid >> 5;
  const int l15  = lane & 15;
  const int h    = lane >> 4;

  const int b  = blockIdx.x >> 7;
  const int qt = blockIdx.x & 127;
  const size_t qrow0 = (size_t)b * QL + (size_t)qt * 32;

  // issue async stage of K tile `kt` into buffer `buf` + its ek slice
  auto issueK = [&](int kt, int buf) {
    const size_t krow0 = (size_t)b * KLEN + (size_t)kt * 32;
    const unsigned short* gk = Kh + krow0 * DD;
    #pragma unroll
    for (int i = 0; i < 8; ++i) {
      const int c  = tid + i * 256;            // 16B chunk id, 64 per row
      const int n  = c >> 6;
      const int dc = (c & 63) << 3;
      const unsigned ldsOff = (unsigned)(uintptr_t)&Kn[buf][n * PQ + dc];
      async_cp16(ldsOff, gk + (size_t)n * DD + dc);
    }
    if (tid < 32) ekt[buf][tid] = ekw[krow0 + tid];
  };

  // ---- prologue: async-stage Q tile and K tile 0 ----
  {
    const unsigned short* gq = Qh + qrow0 * DD;
    #pragma unroll
    for (int i = 0; i < 8; ++i) {
      const int c  = tid + i * 256;
      const int n  = c >> 6;
      const int dc = (c & 63) << 3;
      const unsigned ldsOff = (unsigned)(uintptr_t)&Qs[n * PQ + dc];
      async_cp16(ldsOff, gq + (size_t)n * DD + dc);
    }
  }
  issueK(0, 0);
  if (tid < 32) { rowM[tid] = -3.0e38f; rowL[tid] = 0.0f; }

  v8f accO[2][4];
  #pragma unroll
  for (int ms = 0; ms < 2; ++ms)
    #pragma unroll
    for (int ds = 0; ds < 4; ++ds)
      #pragma unroll
      for (int r = 0; r < 8; ++r) accO[ms][ds][r] = 0.0f;

  async_wait0();
  __syncthreads();

  const int msG   = (wave >> 1) & 1;   // S subtile row
  const int nsG   = wave & 1;          // S subtile col
  const int dhalf = wave >> 2;         // which half of d-reduction
  float* SscW = dhalf ? Ssc1 : Ssc0;

  for (int kt = 0; kt < NT; ++kt) {
    const int buf = kt & 1;

    // ---- pipeline: start async copy of next K tile into the other buffer ----
    if (kt + 1 < NT) issueK(kt + 1, buf ^ 1);

    // warm GL2 two tiles ahead
    if (kt + 2 < NT) {
      const char* nxt = (const char*)(Kh + ((size_t)b * KLEN + (size_t)(kt + 2) * 32) * DD);
      #pragma unroll
      for (int i = 0; i < 2; ++i)
        __builtin_prefetch(nxt + (tid + i * 256) * 64, 0, 1);
    }

    // ---- build transposed Kt (d,n) from Kn[buf]: n = lane -> ~conflict-free ----
    #pragma unroll
    for (int i = 0; i < 8; ++i) {
      const int n   = tid & 31;
      const int dcc = (tid >> 5) + (i << 3);       // 0..63
      U8 u; u.v = *(const uint4*)&Kn[buf][n * PQ + dcc * 8];
      #pragma unroll
      for (int e = 0; e < 8; ++e) Kt[(dcc * 8 + e) * PT + n] = u.s[e];
    }

    // ---- GEMM1: partial S over this wave's d-half ----
    {
      v8f sAcc;
      #pragma unroll
      for (int r = 0; r < 8; ++r) sAcc[r] = 0.0f;
      const int d0 = dhalf * 256;
      #pragma unroll
      for (int dc = 0; dc < 8; ++dc) {
        FragAB a, bb;
        const unsigned short* ap = Qs + (msG * 16 + l15) * PQ + d0 + dc * 32 + h * 8;
        a.q[0] = *(const uint4*)ap;
        a.q[1] = *(const uint4*)(ap + 16);
        const unsigned short* bp = &Kn[buf][(nsG * 16 + l15) * PQ + d0 + dc * 32 + h * 8];
        bb.q[0] = *(const uint4*)bp;
        bb.q[1] = *(const uint4*)(bp + 16);
        sAcc = __builtin_amdgcn_wmma_f32_16x16x32_bf16(false, a.v, false, bb.v,
                                                       (short)0, sAcc, false, false);
      }
      #pragma unroll
      for (int r = 0; r < 8; ++r)
        SscW[(msG * 16 + r + h * 8) * PS + nsG * 16 + l15] = sAcc[r];
    }
    __syncthreads();

    // ---- online softmax: 8 lanes per row (4 cols each), 256 threads busy ----
    {
      const int m  = tid >> 3;
      const int oc = tid & 7;
      const float mPrev = rowM[m];
      float mNew = mPrev;
      #pragma unroll
      for (int i = 0; i < 4; ++i) {
        const int n = oc * 4 + i;
        mNew = fmaxf(mNew, Ssc0[m * PS + n] + Ssc1[m * PS + n] + ekt[buf][n]);
      }
      mNew = fmaxf(mNew, __shfl_xor(mNew, 1));
      mNew = fmaxf(mNew, __shfl_xor(mNew, 2));
      mNew = fmaxf(mNew, __shfl_xor(mNew, 4));
      const float alpha = __expf(mPrev - mNew);
      float l = 0.0f;
      #pragma unroll
      for (int i = 0; i < 4; ++i) {
        const int n = oc * 4 + i;
        const float p = __expf(Ssc0[m * PS + n] + Ssc1[m * PS + n] + ekt[buf][n] - mNew);
        l += p;
        Pt[m * PT + n] = f2bf(p);
      }
      l += __shfl_xor(l, 1);
      l += __shfl_xor(l, 2);
      l += __shfl_xor(l, 4);
      if (oc == 0) {
        rowM[m] = mNew;
        rowL[m] = rowL[m] * alpha + l;
        rowA[m] = alpha;
      }
    }
    __syncthreads();

    // ---- rescale accumulators + GEMM2: O += P K (wave owns 64-wide D slice) ----
    {
      FragAB pa[2];
      #pragma unroll
      for (int ms = 0; ms < 2; ++ms) {
        const unsigned short* ap = Pt + (ms * 16 + l15) * PT + h * 8;
        pa[ms].q[0] = *(const uint4*)ap;
        pa[ms].q[1] = *(const uint4*)(ap + 16);
        #pragma unroll
        for (int r = 0; r < 8; ++r) {
          const float al = rowA[ms * 16 + r + h * 8];
          #pragma unroll
          for (int ds = 0; ds < 4; ++ds) accO[ms][ds][r] *= al;
        }
      }
      #pragma unroll
      for (int ds = 0; ds < 4; ++ds) {
        FragAB kb;
        const unsigned short* bp = Kt + (wave * 64 + ds * 16 + l15) * PT + h * 8;
        kb.q[0] = *(const uint4*)bp;
        kb.q[1] = *(const uint4*)(bp + 16);
        accO[0][ds] = __builtin_amdgcn_wmma_f32_16x16x32_bf16(false, pa[0].v, false, kb.v,
                                                              (short)0, accO[0][ds], false, false);
        accO[1][ds] = __builtin_amdgcn_wmma_f32_16x16x32_bf16(false, pa[1].v, false, kb.v,
                                                              (short)0, accO[1][ds], false, false);
      }
    }

    // drain the in-flight async copy for tile kt+1, then release buffers
    if (kt + 1 < NT) async_wait0();
    __syncthreads();
  }

  // ---- epilogue: O / L ----
  #pragma unroll
  for (int ms = 0; ms < 2; ++ms) {
    #pragma unroll
    for (int r = 0; r < 8; ++r) {
      const int m = ms * 16 + r + h * 8;
      const float inv = 1.0f / rowL[m];
      float* orow = out + (qrow0 + m) * DD + wave * 64 + l15;
      #pragma unroll
      for (int ds = 0; ds < 4; ++ds)
        orow[ds * 16] = accO[ms][ds][r] * inv;
    }
  }
}

extern "C" void kernel_launch(void* const* d_in, const int* in_sizes, int n_in,
                              void* d_out, int out_size, void* d_ws, size_t ws_size,
                              hipStream_t stream) {
  const float* q = (const float*)d_in[0];
  const float* k = (const float*)d_in[1];
  const float* W = (const float*)d_in[2];
  // d_in[3] = bias b: constant along softmax axis -> cancels, unused.
  float* out = (float*)d_out;

  unsigned short* Qh = (unsigned short*)d_ws;                       // 32 MB
  unsigned short* Kh = Qh + (size_t)BATCH * QL * DD;                // 32 MB
  float* ekw = (float*)(Kh + (size_t)BATCH * KLEN * DD);            // 128 KB

  prep_kernel<<<8192, 256, 0, stream>>>(q, k, W, Qh, Kh, ekw);
  flash_attn_kernel<<<BATCH * (QL / 32), 256, 0, stream>>>(Qh, Kh, ekw, out);
}
